// DeformableConvModule_54812372631635
// MI455X (gfx1250) — matrix operand order repaired and checked by reading, not compiled, not run on hardware
//
#include <hip/hip_runtime.h>
#include <hip/hip_bf16.h>
#include <math.h>

// ---------------------------------------------------------------------------
// Deformable conv: conv3x3(96->192) -> per-channel grid_sample -> conv3x3(96->192)
// Convs as implicit GEMM on v_wmma_f32_16x16x32_bf16 (fp32 accumulate).
//   GEMM: M = B*H*W = 200704, N = 192, K = 96*9 = 864 (27 k-steps of 32)
// Block: 256 threads (8 waves). Tile BM=128 x BN=64, BK=32.
// Each wave: 16 M-rows x 64 N-cols = four 16x16 f32 accumulators,
// one A fragment reused across 4 WMMAs per k-step.
// Weights are pre-converted to bf16 once (L2-resident), so B staging is
// conversion-free b32 copies.
// ---------------------------------------------------------------------------

typedef __attribute__((ext_vector_type(16))) __bf16 v16bf;
typedef __attribute__((ext_vector_type(8)))  float  v8f;

#define BDIM  4
#define CIN   96
#define COUT  192
#define HDIM  224
#define WDIM  224
#define HWSZ  (HDIM * WDIM)      // 50176, divisible by BM
#define KTOT  (CIN * 9)          // 864
#define WELEMS (COUT * KTOT)     // 165888
#define BK    32
#define BM    128
#define BN    64
#define NT    256
#define LDA   40                 // padded LDS k-stride (halves): 80B rows, 16B-aligned frags
#define LDB   40

// ---- one-shot fp32 -> bf16 weight conversion (both conv filters) ----------
__global__ void cvt_weights_kernel(const float* __restrict__ w0,
                                   const float* __restrict__ w1,
                                   __bf16* __restrict__ o0,
                                   __bf16* __restrict__ o1)
{
    const int i = blockIdx.x * blockDim.x + threadIdx.x;
    if (i < WELEMS) {
        o0[i] = (__bf16)w0[i];
        o1[i] = (__bf16)w1[i];
    }
}

__global__ __launch_bounds__(NT)
void conv3x3_wmma_kernel(const float* __restrict__ xin,
                         const __bf16* __restrict__ wgt,  // bf16 [COUT][CIN][3][3] == [n][k]
                         const float* __restrict__ bias,  // [COUT]
                         float* __restrict__ out)         // [B][COUT][H][W]
{
    __shared__ __bf16 lA[BM][LDA];   // [m][k]
    __shared__ __bf16 lB[BN][LDB];   // [n][k]

    const int tid    = threadIdx.x;
    const int lane   = tid & 31;
    const int wave   = tid >> 5;
    const int laneHi = (lane >> 4) & 1;
    const int lane16 = lane & 15;

    const int m0 = blockIdx.x * BM;
    const int n0 = blockIdx.y * BN;

    // HWSZ % BM == 0 -> tile lies inside one batch image: b, hw0 are scalars.
    const int bidx  = m0 / HWSZ;
    const int hw0   = m0 - bidx * HWSZ;
    const int bbase = bidx * (CIN * HWSZ);

    // ---- loader geometry: each thread owns 2 adjacent k-cols ----
    const int colp = (tid & 15) * 2;   // even k column
    const int rowb = tid >> 4;         // 0..15; A rows = rowb + 16*i (i<8)

    // per-row pixel coords, hoisted out of the k-loop
    int ph[8], pw[8];
#pragma unroll
    for (int i = 0; i < 8; ++i) {
        const int hw = hw0 + rowb + (i << 4);
        ph[i] = hw / WDIM;
        pw[i] = hw - ph[i] * WDIM;
    }

    // hoisted B-stage / prefetch base (row rowb of the B tile)
    const __bf16* wrow = wgt + (size_t)(n0 + rowb) * KTOT + colp;

    v8f acc[4] = {{}, {}, {}, {}};

    for (int k0 = 0; k0 < KTOT; k0 += BK) {
        // filter coords for this thread's two k columns (only per-k-step math)
        const int ka  = k0 + colp;
        const int ci0 = ka / 9;
        const int rs0 = ka - ci0 * 9;
        const int r0  = rs0 / 3;
        const int s0  = rs0 - r0 * 3;
        const int kb  = ka + 1;
        const int ci1 = kb / 9;
        const int rs1 = kb - ci1 * 9;
        const int r1  = rs1 / 3;
        const int s1  = rs1 - r1 * 3;
        const int cb0 = bbase + ci0 * HWSZ;
        const int cb1 = bbase + ci1 * HWSZ;

        // ---- stage A tile: branchless clamped gather, packed bf16 stores ----
#pragma unroll
        for (int i = 0; i < 8; ++i) {
            const int ih0 = ph[i] + r0 - 1, iw0 = pw[i] + s0 - 1;
            const int ih1 = ph[i] + r1 - 1, iw1 = pw[i] + s1 - 1;
            const bool v0 = ((unsigned)ih0 < (unsigned)HDIM) & ((unsigned)iw0 < (unsigned)WDIM);
            const bool v1 = ((unsigned)ih1 < (unsigned)HDIM) & ((unsigned)iw1 < (unsigned)WDIM);
            const int c0 = min(max(ih0, 0), HDIM - 1) * WDIM + min(max(iw0, 0), WDIM - 1);
            const int c1 = min(max(ih1, 0), HDIM - 1) * WDIM + min(max(iw1, 0), WDIM - 1);
            float f0 = xin[cb0 + c0]; f0 = v0 ? f0 : 0.0f;
            float f1 = xin[cb1 + c1]; f1 = v1 ? f1 : 0.0f;
            union { __bf16 h[2]; unsigned u; } pk;
            pk.h[0] = (__bf16)f0; pk.h[1] = (__bf16)f1;
            *(unsigned*)&lA[rowb + (i << 4)][colp] = pk.u;
        }

        // ---- stage B tile: bf16 weights, pure b32 copies (no conversion) ----
#pragma unroll
        for (int i = 0; i < 4; ++i) {
            const int nr = rowb + (i << 4);            // 0..63
            *(unsigned*)&lB[nr][colp] =
                *(const unsigned*)(wrow + (size_t)(i << 4) * KTOT + k0);
        }

        // prefetch next weight tile (address is a single add off wrow)
        if (k0 + BK < KTOT)
            __builtin_prefetch(wrow + k0 + BK, 0, 1);

        __syncthreads();

        // ---- fragments in contiguous k order (same for A and B) ----
        // A 16x32: lane row M=L%16; elem e -> K = 16*laneHi + e
        const int arow = wave * 16 + lane16;
        const int kA   = laneHi * 16;
        v16bf af;
#pragma unroll
        for (int e = 0; e < 16; ++e) {
            af[e]     = lA[arow][kA + e];
        }
        // B 32x16: lane col N=L%16; lanes0-15 K=0..15, lanes16-31 K=16..31
        const int kB = laneHi * 16;
        v16bf bf0, bf1, bf2, bf3;
#pragma unroll
        for (int e = 0; e < 16; ++e) {
            bf0[e] = lB[lane16     ][kB + e];
            bf1[e] = lB[lane16 + 16][kB + e];
            bf2[e] = lB[lane16 + 32][kB + e];
            bf3[e] = lB[lane16 + 48][kB + e];
        }

        acc[0] = __builtin_amdgcn_wmma_f32_16x16x32_bf16(false, af, false, bf0, (short)0, acc[0], false, false);
        acc[1] = __builtin_amdgcn_wmma_f32_16x16x32_bf16(false, af, false, bf1, (short)0, acc[1], false, false);
        acc[2] = __builtin_amdgcn_wmma_f32_16x16x32_bf16(false, af, false, bf2, (short)0, acc[2], false, false);
        acc[3] = __builtin_amdgcn_wmma_f32_16x16x32_bf16(false, af, false, bf3, (short)0, acc[3], false, false);

        __syncthreads();
    }

    // ---- store: C layout: VGPR i -> M = i (+8 lanes16-31), N = lane%16.
    // The 8 rows are consecutive hw positions -> two 16B stores per acc.
    const int hwst = hw0 + wave * 16 + laneHi * 8;
#pragma unroll
    for (int j = 0; j < 4; ++j) {
        const int n  = n0 + 16 * j + lane16;
        const float bs = bias[n];
        float4 lo, hi;
        lo.x = acc[j][0] + bs; lo.y = acc[j][1] + bs;
        lo.z = acc[j][2] + bs; lo.w = acc[j][3] + bs;
        hi.x = acc[j][4] + bs; hi.y = acc[j][5] + bs;
        hi.z = acc[j][6] + bs; hi.w = acc[j][7] + bs;
        float* p = out + ((size_t)bidx * COUT + n) * HWSZ + hwst;
        *(float4*)(p)     = lo;
        *(float4*)(p + 4) = hi;
    }
}

// ---------------------------------------------------------------------------
// Per-channel bilinear grid sample, zeros padding, align_corners=False.
// grid[b,c,h,w,0] = offset channel 2c (gx); [...,1] = channel 2c+1 (gy).
// ---------------------------------------------------------------------------
__global__ void grid_sample_kernel(const float* __restrict__ x,
                                   const float* __restrict__ off,  // [B][2*CIN][H][W]
                                   float* __restrict__ out)        // [B][CIN][H][W]
{
    const size_t total = (size_t)BDIM * CIN * HWSZ;
    size_t idx = (size_t)blockIdx.x * blockDim.x + threadIdx.x;
    if (idx >= total) return;

    const int hw = (int)(idx % HWSZ);
    const int c  = (int)((idx / HWSZ) % CIN);
    const int b  = (int)(idx / ((size_t)CIN * HWSZ));

    const float gx = off[((size_t)b * (2 * CIN) + 2 * c)     * HWSZ + hw];
    const float gy = off[((size_t)b * (2 * CIN) + 2 * c + 1) * HWSZ + hw];

    const float ix = ((gx + 1.0f) * (float)WDIM - 1.0f) * 0.5f;
    const float iy = ((gy + 1.0f) * (float)HDIM - 1.0f) * 0.5f;

    const float x0f = floorf(ix);
    const float y0f = floorf(iy);
    const int x0 = (int)x0f, y0 = (int)y0f;
    const int x1 = x0 + 1,   y1 = y0 + 1;
    const float wx1 = ix - x0f, wx0 = 1.0f - wx1;
    const float wy1 = iy - y0f, wy0 = 1.0f - wy1;

    const float* xc = x + ((size_t)b * CIN + c) * HWSZ;
    auto samp = [&](int xi, int yi) -> float {
        if (xi < 0 || xi >= WDIM || yi < 0 || yi >= HDIM) return 0.0f;
        return xc[(size_t)yi * WDIM + xi];
    };

    out[idx] = samp(x0, y0) * (wx0 * wy0)
             + samp(x1, y0) * (wx1 * wy0)
             + samp(x0, y1) * (wx0 * wy1)
             + samp(x1, y1) * (wx1 * wy1);
}

// ---------------------------------------------------------------------------
extern "C" void kernel_launch(void* const* d_in, const int* in_sizes, int n_in,
                              void* d_out, int out_size, void* d_ws, size_t ws_size,
                              hipStream_t stream) {
    const float* x     = (const float*)d_in[0];
    const float* w_off = (const float*)d_in[1];
    const float* b_off = (const float*)d_in[2];
    const float* w_def = (const float*)d_in[3];
    const float* b_def = (const float*)d_in[4];
    float* outp = (float*)d_out;

    // d_out is exactly B*192*H*W floats: use it as the offset scratch (conv2
    // fully overwrites it afterwards).
    // d_ws layout: [ sampled: B*CIN*H*W f32 (77 MB) | w_off bf16 | w_def bf16 ]
    float*  offset   = outp;
    float*  sampled  = (float*)d_ws;
    __bf16* w_off_bf = (__bf16*)((char*)d_ws + (size_t)BDIM * CIN * HWSZ * sizeof(float));
    __bf16* w_def_bf = w_off_bf + WELEMS;

    // one-shot weight conversion to bf16 (L2-resident afterwards)
    cvt_weights_kernel<<<(WELEMS + NT - 1) / NT, NT, 0, stream>>>(
        w_off, w_def, w_off_bf, w_def_bf);

    const int MBLKS = (BDIM * HWSZ) / BM;   // 200704 / 128 = 1568
    const int NBLKS = COUT / BN;            // 3
    dim3 cgrid(MBLKS, NBLKS);

    // conv1: x * w_off -> offset (in d_out)
    conv3x3_wmma_kernel<<<cgrid, NT, 0, stream>>>(x, w_off_bf, b_off, offset);

    // grid sample: (x, offset) -> sampled (in d_ws)
    const size_t total = (size_t)BDIM * CIN * HWSZ;
    grid_sample_kernel<<<(unsigned)((total + 255) / 256), 256, 0, stream>>>(
        x, offset, sampled);

    // conv2: sampled * w_def -> d_out (overwrites offset scratch)
    conv3x3_wmma_kernel<<<cgrid, NT, 0, stream>>>(sampled, w_def_bf, b_def, outp);
}
